// decoder_attention_11510512353889
// MI455X (gfx1250) — matrix operand order repaired and checked by reading, not compiled, not run on hardware
//
#include <hip/hip_runtime.h>
#include <stdint.h>

#define BB 64
#define TT 512
#define HH 1024

typedef __attribute__((ext_vector_type(16))) __bf16    v16bf;
typedef __attribute__((ext_vector_type(8)))  float     v8f;
typedef __attribute__((ext_vector_type(4)))  uint32_t  u32x4;
typedef __attribute__((ext_vector_type(2)))  uint32_t  u32x2;

union FragAB { v16bf v; u32x4 q[2]; };
union FragC  { v8f   v; float f[8]; };

// f32 -> bf16 round-to-nearest-even (bit trick; NaN edge ignored)
__device__ __forceinline__ uint32_t pk_bf16(float x, float y) {
    uint32_t a = __float_as_uint(x), b = __float_as_uint(y);
    a = (a + 0x7fffu + ((a >> 16) & 1u)) >> 16;
    b = (b + 0x7fffu + ((b >> 16) & 1u)) >> 16;
    return (a & 0xffffu) | (b << 16);
}

// Branch-free tanh: copysign(1 - 2/(e^{2|x|}+1), x). v_exp + v_rcp, no exec divergence.
__device__ __forceinline__ float fast_tanh(float x) {
    float ax = fabsf(x);
    float e  = __expf(2.0f * ax);              // inf for large ax -> t = 1
    float t  = 1.0f - 2.0f * __builtin_amdgcn_rcpf(e + 1.0f);
    return copysignf(t, x);
}

// CDNA5 async copy global -> LDS (16B per lane), tracked by ASYNCcnt.
__device__ __forceinline__ void async_ld_b128(void* lds_dst, const void* gsrc) {
    uint32_t lds_off = (uint32_t)(uintptr_t)lds_dst;  // low 32 bits of flat LDS addr = LDS offset
    asm volatile("global_load_async_to_lds_b128 %0, %1, off"
                 :: "v"(lds_off), "v"(gsrc) : "memory");
}
__device__ __forceinline__ void wait_async0() {
    asm volatile("s_wait_asynccnt 0" ::: "memory");
}

// ---------------- K0: W_prev f32 -> bf16 ----------------
__global__ __launch_bounds__(256) void k0_cvt(const float* __restrict__ W,
                                              uint16_t* __restrict__ W16) {
    const int idx = blockIdx.x * 256 + threadIdx.x;   // 262144 float4's
    float4 f = ((const float4*)W)[idx];
    u32x2 p; p.x = pk_bf16(f.x, f.y); p.y = pk_bf16(f.z, f.w);
    ((u32x2*)W16)[idx] = p;
}

// ---------------- K1: dec_fea = s_t @ W_s^T + b_s ----------------
__global__ __launch_bounds__(256) void k1_decfea(const float* __restrict__ s_t,
                                                 const float* __restrict__ W_s,
                                                 const float* __restrict__ b_s,
                                                 float* __restrict__ dec_ws) {
    __shared__ float sl[HH];
    const int b = blockIdx.x, tid = threadIdx.x;
#pragma unroll
    for (int i = 0; i < 4; ++i) sl[tid + i * 256] = s_t[(size_t)b * HH + tid + i * 256];
    __syncthreads();
#pragma unroll
    for (int i = 0; i < 4; ++i) {
        const int o = i * 256 + tid;
        const float4* w = (const float4*)(W_s + (size_t)o * HH);
        float acc = b_s[o];
        for (int hq = 0; hq < HH / 4; ++hq) {
            float4 ww = w[hq];
            const float* sp = &sl[hq * 4];
            acc += ww.x * sp[0] + ww.y * sp[1] + ww.z * sp[2] + ww.w * sp[3];
        }
        dec_ws[(size_t)b * HH + o] = acc;
    }
}

// ---------------- K2: scores[b,t] = sum_o tanh(et + dec_fea)*v  (WMMA bf16) ----------------
// Grid: BB * (TT/16) workgroups of 128 threads (4 waves).
// One 16-row t-tile per WG; wave w covers o in [w*256, w*256+256), processed as
// 8 pairs of 16-wide o-tiles so each A fragment feeds two WMMAs.
__global__ __launch_bounds__(128) void k2_scores(const float* __restrict__ prev_s,
                                                 const uint16_t* __restrict__ W16,
                                                 const float* __restrict__ dec_ws,
                                                 const float* __restrict__ vvec,
                                                 float* __restrict__ sc_ws) {
    __shared__ uint16_t Atile[16 * 1032];   // 16 x 1024 bf16, stride padded +8 (bank-conflict free)
    __shared__ float decl[HH];
    __shared__ float vl[HH];
    __shared__ float scpart[4][16];

    const int tid  = threadIdx.x;
    const int wave = tid >> 5;
    const int lane = tid & 31;
    const int b    = blockIdx.x >> 5;
    const int t0   = (blockIdx.x & 31) << 4;

    // async-stage dec_fea[b,:] and v[:] into LDS (32B per thread each)
    {
        const float* dg = dec_ws + (size_t)b * HH + tid * 8;
        async_ld_b128(&decl[tid * 8],     dg);
        async_ld_b128(&decl[tid * 8 + 4], dg + 4);
        const float* vg = vvec + tid * 8;
        async_ld_b128(&vl[tid * 8],     vg);
        async_ld_b128(&vl[tid * 8 + 4], vg + 4);
    }

    // stage + convert A tile: prev_s[b, t0..t0+15, :] f32 -> bf16 in LDS
    {
        const float* Ap = prev_s + ((size_t)b * TT + t0) * HH;
        for (int idx = tid; idx < 16 * (HH / 4); idx += 128) {
            int r = idx >> 8, c4 = idx & 255;
            float4 f = *(const float4*)(Ap + r * HH + c4 * 4);
            u32x2 p; p.x = pk_bf16(f.x, f.y); p.y = pk_bf16(f.z, f.w);
            *(u32x2*)&Atile[r * 1032 + c4 * 4] = p;
        }
    }
    wait_async0();
    __syncthreads();

    // WMMA fragment addressing (16-bit A 16x32 / B 32x16 layouts per CDNA5 ISA):
    const int mrow  = lane & 15;   // A row m, also B column o (local)
    const int khalf = lane >> 4;
    const char* Abase = (const char*)Atile;
    const int aoff0 = (mrow * 1032 + khalf * 8) * 2;  // k chunk 0..7 (or 8..15)
    const int aoff1 = aoff0 + 32;                     // +16 elements

    float score[8];
#pragma unroll
    for (int r = 0; r < 8; ++r) score[r] = 0.f;

    for (int ot = 0; ot < 8; ++ot) {
        const int ob0 = (wave << 8) + (ot << 5);
        const int ob1 = ob0 + 16;
        const uint16_t* Wrow0 = W16 + (size_t)(ob0 + mrow) * HH + khalf * 8;
        const uint16_t* Wrow1 = Wrow0 + 16 * HH;
        if (ot < 7) __builtin_prefetch(Wrow0 + 32 * HH, 0, 0);

        FragC acc0, acc1;
#pragma unroll
        for (int r = 0; r < 8; ++r) { acc0.f[r] = 0.f; acc1.f[r] = 0.f; }

#pragma unroll 2
        for (int kb = 0; kb < HH; kb += 32) {
            FragAB a, b0, b1;
            a.q[0] = *(const u32x4*)(Abase + aoff0 + kb * 2);
            a.q[1] = *(const u32x4*)(Abase + aoff1 + kb * 2);
            const u32x4* wp0 = (const u32x4*)(Wrow0 + kb);
            const u32x4* wp1 = (const u32x4*)(Wrow1 + kb);
            b0.q[0] = wp0[0]; b0.q[1] = wp0[2];
            b1.q[0] = wp1[0]; b1.q[1] = wp1[2];
            acc0.v = __builtin_amdgcn_wmma_f32_16x16x32_bf16(
                false, a.v, false, b0.v, (short)0, acc0.v, false, false);
            acc1.v = __builtin_amdgcn_wmma_f32_16x16x32_bf16(
                false, a.v, false, b1.v, (short)0, acc1.v, false, false);
        }
        const float df0 = decl[ob0 + mrow], vs0 = vl[ob0 + mrow];
        const float df1 = decl[ob1 + mrow], vs1 = vl[ob1 + mrow];
#pragma unroll
        for (int r = 0; r < 8; ++r) {
            score[r] += fast_tanh(acc0.f[r] + df0) * vs0
                      + fast_tanh(acc1.f[r] + df1) * vs1;
        }
    }

    // C layout: VGPR r -> M = r + khalf*8, N = mrow. Reduce over the 16-lane N group.
#pragma unroll
    for (int r = 0; r < 8; ++r) {
        float s = score[r];
        s += __shfl_xor(s, 1, 16);
        s += __shfl_xor(s, 2, 16);
        s += __shfl_xor(s, 4, 16);
        s += __shfl_xor(s, 8, 16);
        score[r] = s;
    }
    if (mrow == 0) {
#pragma unroll
        for (int r = 0; r < 8; ++r) scpart[wave][khalf * 8 + r] = score[r];
    }
    __syncthreads();
    if (tid < 16) {
        float s = scpart[0][tid] + scpart[1][tid] + scpart[2][tid] + scpart[3][tid];
        sc_ws[(size_t)b * TT + t0 + tid] = s;
    }
}

// ---------------- K3: softmax over T per batch row ----------------
__global__ __launch_bounds__(256) void k3_softmax(const float* __restrict__ sc,
                                                  float* __restrict__ at) {
    __shared__ float red[256];
    const int b = blockIdx.x, tid = threadIdx.x;
    float s0 = sc[(size_t)b * TT + tid];
    float s1 = sc[(size_t)b * TT + 256 + tid];
    red[tid] = fmaxf(s0, s1);
    __syncthreads();
    for (int off = 128; off > 0; off >>= 1) {
        if (tid < off) red[tid] = fmaxf(red[tid], red[tid + off]);
        __syncthreads();
    }
    float M = red[0];
    __syncthreads();
    float e0 = expf(s0 - M), e1 = expf(s1 - M);
    red[tid] = e0 + e1;
    __syncthreads();
    for (int off = 128; off > 0; off >>= 1) {
        if (tid < off) red[tid] += red[tid + off];
        __syncthreads();
    }
    float inv = 1.0f / red[0];
    at[(size_t)b * TT + tid]       = e0 * inv;
    at[(size_t)b * TT + 256 + tid] = e1 * inv;
}

// ---------------- K4: ct_d + concat (single pass over prev_s) ----------------
__global__ __launch_bounds__(256) void k4_out(const float* __restrict__ prev_s,
                                              const float* __restrict__ s_t,
                                              const float* __restrict__ at_ws,
                                              float* __restrict__ out_ct,
                                              float* __restrict__ out_ps) {
    __shared__ float atl[TT];
    const int tid = threadIdx.x;
    const int b = blockIdx.x >> 2;
    const int h = ((blockIdx.x & 3) << 8) + tid;
    if (tid < 128) async_ld_b128(&atl[tid * 4], at_ws + (size_t)b * TT + tid * 4);
    wait_async0();
    __syncthreads();

    const float* ps = prev_s + (size_t)b * TT * HH + h;
    float* o2 = out_ps + (size_t)b * (TT + 1) * HH + h;
    float ct = 0.f;
#pragma unroll 4
    for (int t = 0; t < TT; ++t) {
        float val = ps[(size_t)t * HH];
        ct += atl[t] * val;
        o2[(size_t)t * HH] = val;
    }
    o2[(size_t)TT * HH] = s_t[(size_t)b * HH + h];
    out_ct[(size_t)b * HH + h] = ct;
}

extern "C" void kernel_launch(void* const* d_in, const int* in_sizes, int n_in,
                              void* d_out, int out_size, void* d_ws, size_t ws_size,
                              hipStream_t stream) {
    (void)in_sizes; (void)n_in; (void)out_size; (void)ws_size;
    const float* s_t    = (const float*)d_in[0];
    const float* prev_s = (const float*)d_in[1];
    const float* W_prev = (const float*)d_in[2];
    const float* W_s    = (const float*)d_in[3];
    const float* b_s    = (const float*)d_in[4];
    const float* v      = (const float*)d_in[5];

    // workspace layout: [W16 bf16 2MB][dec_fea 256KB][scores 128KB][at 128KB]
    uint16_t* W16 = (uint16_t*)d_ws;
    float* dec_ws = (float*)((char*)d_ws + (size_t)(2u << 20));
    float* sc_ws  = dec_ws + (size_t)BB * HH;
    float* at_ws  = sc_ws + (size_t)BB * TT;

    float* out_ct = (float*)d_out;
    float* out_ps = out_ct + (size_t)BB * HH;

    k0_cvt   <<<dim3((HH * HH) / (4 * 256)), dim3(256), 0, stream>>>(W_prev, W16);
    k1_decfea<<<dim3(BB),                    dim3(256), 0, stream>>>(s_t, W_s, b_s, dec_ws);
    k2_scores<<<dim3(BB * (TT / 16)),        dim3(128), 0, stream>>>(prev_s, W16, dec_ws, v, sc_ws);
    k3_softmax<<<dim3(BB),                   dim3(256), 0, stream>>>(sc_ws, at_ws);
    k4_out   <<<dim3(BB * 4),                dim3(256), 0, stream>>>(prev_s, s_t, at_ws, out_ct, out_ps);
}
